// BilinearAttention_74380243632828
// MI455X (gfx1250) — matrix operand order repaired and checked by reading, not compiled, not run on hardware
//
#include <hip/hip_runtime.h>
#include <hip/hip_bf16.h>
#include <math.h>

typedef __attribute__((ext_vector_type(16))) _Float16 v16h;
typedef __attribute__((ext_vector_type(8)))  float    v8f;

#define BS   8
#define DDIM 512
#define HH   8
#define DH   64
#define LA   256
#define LB   256

// gfx1250 native transcendental tanh (TRANS op, listed in CDNA5 ISA).
__device__ __forceinline__ float tanh_fast(float x) {
    float y;
    asm("v_tanh_f32 %0, %1" : "=v"(y) : "v"(x));
    return y;
}

// Y = relu(X @ W^T + bias), X:[M,K] row-major, W:[N,K] row-major, Y:[M,N]
// One 16x16 output tile per wave32, K-stepped by 32 with v_wmma_f32_16x16x32_f16.
__global__ __launch_bounds__(256) void gemm_relu_wmma(
    const float* __restrict__ X, const float* __restrict__ W,
    const float* __restrict__ bias, float* __restrict__ Y,
    int M, int N, int K)
{
    const int lane   = threadIdx.x & 31;
    const int wave   = threadIdx.x >> 5;
    const int w      = blockIdx.x * (blockDim.x >> 5) + wave;
    const int tilesN = N >> 4;
    const int tileM  = w / tilesN;
    const int tileN  = w % tilesN;
    if (tileM * 16 >= M) return;

    const int m      = lane & 15;       // A-row (M) / B-col (N) for this lane
    const int hi     = lane >> 4;       // lane[4]
    const int kbA    = hi * 8;          // A frag K-base per ISA layout
    const int kbB    = hi * 8;          // B frag K-base per ISA layout
    const float* xrow = X + (size_t)(tileM * 16 + m) * K;
    const float* wrow = W + (size_t)(tileN * 16 + m) * K;

    v8f acc = {};
    for (int k = 0; k < K; k += 32) {
        v16h af, bf;
#pragma unroll
        for (int j = 0; j < 8; ++j) {
            af[j]     = (_Float16)xrow[k + kbA + j];        // K = kbA + j
            af[j + 8] = (_Float16)xrow[k + kbA + 16 + j];   // K = kbA + 16 + j
        }
#pragma unroll
        for (int j = 0; j < 16; ++j) {
            bf[j] = (_Float16)wrow[k + kbB + (j < 8 ? j : 8 + j)];  // B[k][n] = W[n,k]
        }
        acc = __builtin_amdgcn_wmma_f32_16x16x32_f16(
            false, af, false, bf, (short)0, acc, false, false);
    }

    const int col = tileN * 16 + m;
    const float bv = bias[col];
#pragma unroll
    for (int r = 0; r < 8; ++r) {
        const int row = tileM * 16 + r + hi * 8;            // C: VGPR r -> M=r+8*lane[4]
        float v = acc[r] + bv;
        Y[(size_t)row * N + col] = v > 0.f ? v : 0.f;
    }
}

// Per (b,h): att[i,j] = sum_c tanh(A1[i,c]*B1[j,c])*q[c]; emit row/col means.
// Tiles staged in LDS via gfx1250 async global->LDS copies (ASYNCcnt).
__global__ __launch_bounds__(256) void att_kernel(
    const float* __restrict__ A1, const float* __restrict__ B1,
    const float* __restrict__ q,
    float* __restrict__ rowmean, float* __restrict__ colmean)
{
    __shared__ float sA[LA * DH];          // 64 KB
    __shared__ float sB[LB * (DH + 1)];    // padded: kill 64-bank conflicts
    __shared__ float sq[DH];
    __shared__ float rowsum[LA];

    const int bh = blockIdx.x;
    const int t  = threadIdx.x;
    const size_t base = (size_t)bh * LA * DH;

    const float* gA = A1 + base;
    const float* gB = B1 + base;
    // addrspace(3) offset = low 32 bits of the generic shared-pointer value
    const unsigned ldsA = (unsigned)(unsigned long long)(void*)sA;
    const unsigned ldsB = (unsigned)(unsigned long long)(void*)sB;

    // sA: 4096 16-byte chunks, contiguous in LDS.
    for (int x = t; x < (LA * DH) / 4; x += 256) {
        unsigned loff = ldsA + (unsigned)x * 16u;
        unsigned goff = (unsigned)x * 16u;
        asm volatile("global_load_async_to_lds_b128 %0, %1, %2"
                     :: "v"(loff), "v"(goff), "s"(gA) : "memory");
    }
    // sB: 16 chunks per 64-float row; LDS row stride 65 floats (260 B).
    for (int x = t; x < (LB * DH) / 4; x += 256) {
        unsigned row    = (unsigned)x >> 4;
        unsigned within = ((unsigned)x & 15u) * 16u;
        unsigned loff = ldsB + row * ((DH + 1) * 4u) + within;
        unsigned goff = (unsigned)x * 16u;
        asm volatile("global_load_async_to_lds_b128 %0, %1, %2"
                     :: "v"(loff), "v"(goff), "s"(gB) : "memory");
    }
    if (t < DH) sq[t] = q[t];
    rowsum[t] = 0.f;
    asm volatile("s_wait_asynccnt 0x0" ::: "memory");
    __syncthreads();

    const int j = t;                        // this thread owns column j
    const float* bj = &sB[j * (DH + 1)];
    float colacc = 0.f;
    for (int i = 0; i < LA; ++i) {
        const float* ai = &sA[i * DH];      // broadcast across wave
        float acc = 0.f;
#pragma unroll 4
        for (int c = 0; c < DH; ++c)
            acc += tanh_fast(ai[c] * bj[c]) * sq[c];
        colacc += acc;
        float r = acc;                      // wave reduce, then 1 atomic/wave
#pragma unroll
        for (int s = 16; s > 0; s >>= 1) r += __shfl_down(r, s, 32);
        if ((t & 31) == 0) atomicAdd(&rowsum[i], r);
    }
    colmean[bh * LB + j] = colacc * (1.f / LA);
    __syncthreads();
    rowmean[bh * LA + t] = rowsum[t] * (1.f / LB);
}

// Softmax over 256-length rows. grid = #rows, block = 256.
__global__ __launch_bounds__(256) void softmax256(
    const float* __restrict__ src, float* __restrict__ dst)
{
    __shared__ float red[256];
    const int r = blockIdx.x, t = threadIdx.x;
    const float v = src[r * 256 + t];
    red[t] = v; __syncthreads();
#pragma unroll
    for (int s = 128; s > 0; s >>= 1) {
        if (t < s) red[t] = fmaxf(red[t], red[t + s]);
        __syncthreads();
    }
    const float mx = red[0]; __syncthreads();
    const float e = __expf(v - mx);
    red[t] = e; __syncthreads();
#pragma unroll
    for (int s = 128; s > 0; s >>= 1) {
        if (t < s) red[t] += red[t + s];
        __syncthreads();
    }
    dst[r * 256 + t] = e / red[0];
}

// A_p / B_p weighted pooling of the ORIGINAL inputs (raw reshape => flat index).
__global__ __launch_bounds__(128) void pool_kernel(
    const float* __restrict__ A, const float* __restrict__ B,
    const float* __restrict__ tb2a, const float* __restrict__ ta2b,
    float* __restrict__ out)
{
    const int bh = blockIdx.x;              // 0..63
    const int b = bh >> 3, h = bh & 7;
    const int t = threadIdx.x;
    const int side = t >> 6;                // 0: A_p, 1: B_p
    const int c = t & 63;
    const float* X = side ? B : A;
    const float* w = side ? ta2b : tb2a;
    const size_t base = (size_t)bh * LA * DH;
    float acc = 0.f;
    for (int i = 0; i < LA; ++i)
        acc += X[base + (size_t)i * DH + c] * w[bh * LA + i];
    out[b * 1024 + side * 512 + h * 64 + c] = acc;
}

// Mean over heads of temp_b2a / temp_a2b -> outputs 2 and 3.
__global__ __launch_bounds__(256) void meanh_kernel(
    const float* __restrict__ tb2a, const float* __restrict__ ta2b,
    float* __restrict__ out)
{
    const int b = blockIdx.x, t = threadIdx.x;
    float s1 = 0.f, s2 = 0.f;
#pragma unroll
    for (int h = 0; h < HH; ++h) {
        s1 += tb2a[(b * HH + h) * 256 + t];
        s2 += ta2b[(b * HH + h) * 256 + t];
    }
    out[8192 + b * 256 + t]        = s1 * 0.125f;
    out[8192 + 2048 + b * 256 + t] = s2 * 0.125f;
}

extern "C" void kernel_launch(void* const* d_in, const int* in_sizes, int n_in,
                              void* d_out, int out_size, void* d_ws, size_t ws_size,
                              hipStream_t stream) {
    const float* A  = (const float*)d_in[0];
    const float* B  = (const float*)d_in[1];
    const float* W1 = (const float*)d_in[2];
    const float* b1 = (const float*)d_in[3];
    const float* W2 = (const float*)d_in[4];
    const float* b2 = (const float*)d_in[5];
    const float* q  = (const float*)d_in[6];
    float* out = (float*)d_out;

    float* ws      = (float*)d_ws;
    float* A1      = ws;                 // 8*256*512 = 1048576 f32
    float* B1      = A1 + 1048576;       // 1048576
    float* rowmean = B1 + 1048576;       // 64*256
    float* colmean = rowmean + 16384;
    float* tb2a    = colmean + 16384;
    float* ta2b    = tb2a + 16384;

    // (2048/16)*(512/16) = 4096 wave-tiles, 8 waves/block -> 512 blocks
    gemm_relu_wmma<<<512, 256, 0, stream>>>(A, W1, b1, A1, BS * LA, DDIM, DDIM);
    gemm_relu_wmma<<<512, 256, 0, stream>>>(B, W2, b2, B1, BS * LB, DDIM, DDIM);
    att_kernel<<<BS * HH, 256, 0, stream>>>(A1, B1, q, rowmean, colmean);
    softmax256<<<BS * HH, 256, 0, stream>>>(rowmean, tb2a);
    softmax256<<<BS * HH, 256, 0, stream>>>(colmean, ta2b);
    pool_kernel<<<BS * HH, 128, 0, stream>>>(A, B, tb2a, ta2b, out);
    meanh_kernel<<<BS, 256, 0, stream>>>(tb2a, ta2b, out);
}
